// FiLM_38766374814033
// MI455X (gfx1250) — matrix operand order repaired and checked by reading, not compiled, not run on hardware
//
#include <hip/hip_runtime.h>
#include <cstdint>
#include <cstddef>

// ---------------------------------------------------------------------------
// FiLM forecaster, collapsed-linear formulation for MI455X (gfx1250, wave32).
// Heavy math on V_WMMA_F32_16X16X32_BF16 (bf16x3 splitting, f32 accumulate).
// H-GEMM tiles staged into LDS with the Tensor Data Mover (TENSORcnt waits).
// ---------------------------------------------------------------------------

typedef __bf16 bf16;
typedef __bf16 v8bf  __attribute__((ext_vector_type(8)));
typedef __bf16 v16bf __attribute__((ext_vector_type(16)));
typedef float  v8f   __attribute__((ext_vector_type(8)));
typedef unsigned int u32x4 __attribute__((ext_vector_type(4)));
typedef int i32x4 __attribute__((ext_vector_type(4)));
typedef int i32x8 __attribute__((ext_vector_type(8)));

#define SEQ    384
#define PRED   96
#define NHIP   256
#define MODES_ 32
#define KDIM_H 16384      // 2 * MODES_ * NHIP  (R|S concatenated)
#define BE     512        // BATCH * ENC_IN

#ifndef USE_TDM
#define USE_TDM 1
#endif
#if USE_TDM && defined(__has_builtin)
#if __has_builtin(__builtin_amdgcn_tensor_load_to_lds)
#define HAVE_TDM 1
#endif
#endif
#ifndef HAVE_TDM
#define HAVE_TDM 0
#endif

// ---------------------------- utility kernels ------------------------------

__global__ void k_zero_f32(float* __restrict__ p, int n) {
  int i = blockIdx.x * blockDim.x + threadIdx.x;
  if (i < n) p[i] = 0.0f;
}

__global__ void k_split_bf16(const float* __restrict__ w, bf16* __restrict__ hi,
                             bf16* __restrict__ lo, int n) {
  int i = blockIdx.x * blockDim.x + threadIdx.x;
  if (i < n) {
    float v = w[i];
    bf16 h = (bf16)v;
    hi[i] = h;
    lo[i] = (bf16)(v - (float)h);
  }
}

// affine + transpose (b,t,e)->(be,t) + bf16 hi/lo split
__global__ void k_affine(const float* __restrict__ x, const float* __restrict__ aw,
                         const float* __restrict__ ab, bf16* __restrict__ XThi,
                         bf16* __restrict__ XTlo) {
  int b = blockIdx.x, t = blockIdx.y, e = threadIdx.x;   // 8 x 384 blocks, 64 thr
  float v = x[(b * SEQ + t) * 64 + e] * aw[e] + ab[e];
  bf16 h = (bf16)v;
  int idx = (b * 64 + e) * SEQ + t;
  XThi[idx] = h;
  XTlo[idx] = (bf16)(v - (float)h);
}

// ------------------- K[d] = A^d B  (sequential matvec chain) ----------------

__global__ __launch_bounds__(1024)
void k_hippoK(const float* __restrict__ A, const float* __restrict__ Bv,
              float* __restrict__ Kb, int L) {
  __shared__ float prev[NHIP];
  __shared__ float part[1024];
  int tid = threadIdx.x;
  int i = tid >> 2, q = tid & 3;      // 4 threads per output row
  if (tid < NHIP) { prev[tid] = Bv[tid]; Kb[tid] = Bv[tid]; }  // K[0] = B
  __syncthreads();
  for (int t = 1; t < L; ++t) {
    const float* Ar = A + i * NHIP + q * 64;
    const float* pv = prev + q * 64;
    float acc = 0.0f;
#pragma unroll 8
    for (int j = 0; j < 64; ++j) acc = fmaf(Ar[j], pv[j], acc);
    part[tid] = acc;
    __syncthreads();
    if (tid < NHIP) {
      float sv = part[tid * 4] + part[tid * 4 + 1] + part[tid * 4 + 2] + part[tid * 4 + 3];
      Kb[t * NHIP + tid] = sv;
      prev[tid] = sv;
    }
    __syncthreads();
  }
}

// ------- R/S: truncated-Fourier correlations of K, written as GEMM-A --------

__global__ __launch_bounds__(256)
void k_RS(const float* __restrict__ Kb, bf16* __restrict__ Ahi,
          bf16* __restrict__ Alo, int L) {
  __shared__ float Kt[32 * NHIP];   // 32 KB
  __shared__ float cwt[32 * 32];    // [k][d] 4 KB
  __shared__ float swt[32 * 32];    // 4 KB
  int s = blockIdx.x;               // 0..L-1
  int tid = threadIdx.x;            // i = 0..255
  float accr[MODES_], accs[MODES_];
#pragma unroll
  for (int k = 0; k < MODES_; ++k) { accr[k] = 0.0f; accs[k] = 0.0f; }

  for (int d0 = 0; s + d0 < L; d0 += 32) {
    int dmax = L - s - d0; if (dmax > 32) dmax = 32;
    for (int x = tid; x < dmax * NHIP; x += 256) Kt[x] = Kb[d0 * NHIP + x];
    for (int x = tid; x < 32 * 32; x += 256) {
      int k = x >> 5, d = x & 31;
      int t = s + d0 + d;
      float sc = (k == 0 ? 1.0f : 2.0f) / (float)L;
      int u = ((k * (t - 95)) % L + L) % L;    // exact phase reduction
      float ang = 6.28318530717958647f * (float)u / (float)L;
      cwt[x] = sc * cosf(ang);
      swt[x] = sc * sinf(ang);
    }
    __syncthreads();
    for (int d = 0; d < dmax; ++d) {
      float kv = Kt[d * NHIP + tid];
#pragma unroll
      for (int k = 0; k < MODES_; ++k) {
        accr[k] = fmaf(kv, cwt[k * 32 + d], accr[k]);
        accs[k] = fmaf(kv, swt[k * 32 + d], accs[k]);
      }
    }
    __syncthreads();
  }

  size_t base = (size_t)s * KDIM_H;
#pragma unroll
  for (int k = 0; k < MODES_; ++k) {
    float r = accr[k];
    bf16 rh = (bf16)r;
    Ahi[base + k * NHIP + tid] = rh;
    Alo[base + k * NHIP + tid] = (bf16)(r - (float)rh);
    float v = accs[k];
    bf16 vh = (bf16)v;
    Ahi[base + 8192 + k * NHIP + tid] = vh;
    Alo[base + 8192 + k * NHIP + tid] = (bf16)(v - (float)vh);
  }
}

// ------- pack wr/wi (i,o,k) -> B[N=o][K = k*256+i | 8192 + k*256+i] ---------

__global__ void k_packW(const float* __restrict__ wr, const float* __restrict__ wi,
                        bf16* __restrict__ Bhi, bf16* __restrict__ Blo) {
  int o = blockIdx.x;      // 0..255
  int tid = threadIdx.x;   // 0..255
  for (int c = 0; c < 64; ++c) {
    int ki = c * 256 + tid;           // 0..16383
    int kk = ki & 8191;
    int k = kk >> 8, i = kk & 255;
    const float* src = (ki < 8192) ? wr : wi;
    float v = src[(i * 256 + o) * MODES_ + k];
    bf16 h = (bf16)v;
    size_t idx = (size_t)o * KDIM_H + ki;
    Bhi[idx] = h;
    Blo[idx] = (bf16)(v - (float)h);
  }
}

// ------------------------- WMMA fragment helpers ---------------------------
// ISA layouts (CDNA5 ISA):
//   A 16x32 bf16 : lane<16 -> M=lane, K in {0..7, 16..23}; lane>=16 -> {8..15, 24..31}
//   B 32x16 bf16 : lane<16 -> N=lane, K in {0..7, 16..23}; lane>=16 -> {8..15, 24..31}
//   C/D f32      : VGPR r -> M = 8*(lane>>4) + r, N = lane&15

__device__ __forceinline__ v16bf ld_frag16(const bf16* p0, const bf16* p1) {
  v8bf c0 = *(const v8bf*)p0;
  v8bf c1 = *(const v8bf*)p1;
  v16bf r;
#pragma unroll
  for (int q = 0; q < 8; ++q) { r[q] = c0[q]; r[8 + q] = c1[q]; }
  return r;
}

// --------------------- TDM 2-D tile load descriptor -------------------------
// D# packing per CDNA5 ISA (group0: count/lds/global/type,
// group1: data_size=2B, tensor dims (oversized: accesses are in-bounds),
// tile_dim0 = width elems, tile_dim1 = rows, 48-bit row stride).

#if HAVE_TDM
__device__ __forceinline__ void tdm_load_2d(unsigned lds_off, const void* gptr,
                                            unsigned rows, unsigned width_elems,
                                            unsigned long long row_stride_elems) {
  unsigned long long ga = (unsigned long long)(uintptr_t)gptr;
  u32x4 g0;
  g0[0] = 1u;                                            // count=1, user mode
  g0[1] = lds_off;                                       // LDS byte address
  g0[2] = (unsigned)(ga & 0xFFFFFFFFu);                  // global_addr[31:0]
  g0[3] = (unsigned)((ga >> 32) & 0x1FFFFFFu) | (2u << 30);  // [56:32] | type=2
  const unsigned td0 = 0x7FFFFFFFu, td1 = 0x7FFFFFFFu;   // huge tensor extents
  i32x8 g1;
  g1[0] = (int)(1u << 16);                               // data_size = 2 bytes
  g1[1] = (int)((td0 & 0xFFFFu) << 16);                  // tensor_dim0 lo
  g1[2] = (int)(((td0 >> 16) & 0xFFFFu) | ((td1 & 0xFFFFu) << 16));
  g1[3] = (int)(((td1 >> 16) & 0xFFFFu) | ((width_elems & 0xFFFFu) << 16));
  g1[4] = (int)(rows & 0xFFFFu);                         // tile_dim1 (tile_dim2=0)
  g1[5] = (int)(unsigned)(row_stride_elems & 0xFFFFFFFFull);
  g1[6] = (int)(unsigned)((row_stride_elems >> 32) & 0xFFFFull);
  g1[7] = 0;
  i32x4 z4; z4[0] = 0; z4[1] = 0; z4[2] = 0; z4[3] = 0;
#if __clang_major__ >= 23
  i32x8 z8;
#pragma unroll
  for (int q = 0; q < 8; ++q) z8[q] = 0;
  __builtin_amdgcn_tensor_load_to_lds(g0, g1, z4, z4, z8, 0);
#else
  __builtin_amdgcn_tensor_load_to_lds(g0, g1, z4, z4, 0);
#endif
}
#endif

// ------------- H-GEMM: 4 waves / WG, 32x32 tile, LDS double buffer ----------
// A: [M][K] bf16 hi/lo, B: [N][K] bf16 hi/lo, C: f32 [M][256]; K = 16384.

__global__ __launch_bounds__(128)
void k_gemmH_lds(const bf16* __restrict__ Ahi, const bf16* __restrict__ Alo,
                 const bf16* __restrict__ Bhi, const bf16* __restrict__ Blo,
                 float* __restrict__ C, int Kdim) {
  __shared__ bf16 sA[2][2][32 * 32];   // [buf][hi/lo][row*32+k] : 8 KB
  __shared__ bf16 sB[2][2][32 * 32];   // 8 KB
  int tid = threadIdx.x;
  int lane = tid & 31, wave = tid >> 5;
  int half = lane >> 4, l16 = lane & 15;
  int wy = wave >> 1, wx = wave & 1;       // 2x2 waves -> 32x32 tile
  int m0 = blockIdx.y * 32, n0 = blockIdx.x * 32;
  const int nk = Kdim / 32;

#if HAVE_TDM
  bool issuer = (wave == 0);
  if (issuer) {
    tdm_load_2d((unsigned)(uintptr_t)&sA[0][0][0], Ahi + (size_t)m0 * Kdim, 32, 32, Kdim);
    tdm_load_2d((unsigned)(uintptr_t)&sA[0][1][0], Alo + (size_t)m0 * Kdim, 32, 32, Kdim);
    tdm_load_2d((unsigned)(uintptr_t)&sB[0][0][0], Bhi + (size_t)n0 * Kdim, 32, 32, Kdim);
    tdm_load_2d((unsigned)(uintptr_t)&sB[0][1][0], Blo + (size_t)n0 * Kdim, 32, 32, Kdim);
  }
#endif

  v8f c = {0.f, 0.f, 0.f, 0.f, 0.f, 0.f, 0.f, 0.f};
  for (int i = 0; i < nk; ++i) {
    int buf = i & 1;
#if HAVE_TDM
    if (issuer) {
      if (i + 1 < nk) {
        int kb = (i + 1) * 32, nb = buf ^ 1;
        tdm_load_2d((unsigned)(uintptr_t)&sA[nb][0][0], Ahi + (size_t)m0 * Kdim + kb, 32, 32, Kdim);
        tdm_load_2d((unsigned)(uintptr_t)&sA[nb][1][0], Alo + (size_t)m0 * Kdim + kb, 32, 32, Kdim);
        tdm_load_2d((unsigned)(uintptr_t)&sB[nb][0][0], Bhi + (size_t)n0 * Kdim + kb, 32, 32, Kdim);
        tdm_load_2d((unsigned)(uintptr_t)&sB[nb][1][0], Blo + (size_t)n0 * Kdim + kb, 32, 32, Kdim);
        __builtin_amdgcn_s_wait_tensorcnt(4);   // stage i complete (in-order)
      } else {
        __builtin_amdgcn_s_wait_tensorcnt(0);
      }
    }
    __syncthreads();                            // publish stage i LDS to all waves
#else
    // fallback: cooperative global->LDS copy (single buffer semantics)
    {
      int kb = i * 32;
      const bf16* gA0 = Ahi + (size_t)m0 * Kdim + kb;
      const bf16* gA1 = Alo + (size_t)m0 * Kdim + kb;
      const bf16* gB0 = Bhi + (size_t)n0 * Kdim + kb;
      const bf16* gB1 = Blo + (size_t)n0 * Kdim + kb;
      int row = tid >> 2, seg = (tid & 3) * 8;  // 128 thr * 8 elems = 1024
      *(v8bf*)&sA[buf][0][row * 32 + seg] = *(const v8bf*)(gA0 + (size_t)row * Kdim + seg);
      *(v8bf*)&sA[buf][1][row * 32 + seg] = *(const v8bf*)(gA1 + (size_t)row * Kdim + seg);
      *(v8bf*)&sB[buf][0][row * 32 + seg] = *(const v8bf*)(gB0 + (size_t)row * Kdim + seg);
      *(v8bf*)&sB[buf][1][row * 32 + seg] = *(const v8bf*)(gB1 + (size_t)row * Kdim + seg);
      __syncthreads();
    }
#endif
    const bf16* pAh = &sA[buf][0][(wy * 16 + l16) * 32 + half * 8];
    const bf16* pAl = &sA[buf][1][(wy * 16 + l16) * 32 + half * 8];
    const bf16* pBh = &sB[buf][0][(wx * 16 + l16) * 32 + half * 8];
    const bf16* pBl = &sB[buf][1][(wx * 16 + l16) * 32 + half * 8];
    v16bf ah = ld_frag16(pAh, pAh + 16);
    v16bf al = ld_frag16(pAl, pAl + 16);
    v16bf bh = ld_frag16(pBh, pBh + 16);
    v16bf bl = ld_frag16(pBl, pBl + 16);
    c = __builtin_amdgcn_wmma_f32_16x16x32_bf16(false, ah, false, bh, (short)0, c, false, false);
    c = __builtin_amdgcn_wmma_f32_16x16x32_bf16(false, ah, false, bl, (short)0, c, false, false);
    c = __builtin_amdgcn_wmma_f32_16x16x32_bf16(false, al, false, bh, (short)0, c, false, false);
    __syncthreads();                            // done reading before overwrite
  }

  int n = n0 + wx * 16 + l16;
#pragma unroll
  for (int r = 0; r < 8; ++r) {
    int m = m0 + wy * 16 + half * 8 + r;
    C[(size_t)m * NHIP + n] = c[r];
  }
}

// --------------- direct-load bf16x3 GEMM (final small GEMM) ----------------

__global__ __launch_bounds__(32)
void k_gemm_bf16x3(const bf16* __restrict__ Ahi, const bf16* __restrict__ Alo,
                   const bf16* __restrict__ Bhi, const bf16* __restrict__ Blo,
                   float* __restrict__ C, int Kdim, int Nld,
                   const float* __restrict__ bias, int has_bias) {
  int lane = threadIdx.x;
  int half = lane >> 4;
  int l16  = lane & 15;
  int rowA = blockIdx.y * 16 + l16;    // M
  int rowB = blockIdx.x * 16 + l16;    // N
  const bf16* aPh = Ahi + (size_t)rowA * Kdim;
  const bf16* aPl = Alo + (size_t)rowA * Kdim;
  const bf16* bPh = Bhi + (size_t)rowB * Kdim;
  const bf16* bPl = Blo + (size_t)rowB * Kdim;

  v8f c = {0.f, 0.f, 0.f, 0.f, 0.f, 0.f, 0.f, 0.f};
  for (int kb = 0; kb < Kdim; kb += 32) {
    int ao = kb + half * 8;
    int bo = kb + half * 8;
    v16bf ah = ld_frag16(aPh + ao, aPh + ao + 16);
    v16bf al = ld_frag16(aPl + ao, aPl + ao + 16);
    v16bf bh = ld_frag16(bPh + bo, bPh + bo + 16);
    v16bf bl = ld_frag16(bPl + bo, bPl + bo + 16);
    c = __builtin_amdgcn_wmma_f32_16x16x32_bf16(false, ah, false, bh, (short)0, c, false, false);
    c = __builtin_amdgcn_wmma_f32_16x16x32_bf16(false, ah, false, bl, (short)0, c, false, false);
    c = __builtin_amdgcn_wmma_f32_16x16x32_bf16(false, al, false, bh, (short)0, c, false, false);
  }
  float badd = has_bias ? bias[0] : 0.0f;
  int n = blockIdx.x * 16 + l16;
#pragma unroll
  for (int r = 0; r < 8; ++r) {
    int m = blockIdx.y * 16 + half * 8 + r;
    C[(size_t)m * Nld + n] = c[r] + badd;
  }
}

// ---- fold em + proj1_w:  WtotT[p, 384-L+s] += p1w[m] * sum_o H[s,o]*em[L-96+p,o]

__global__ void k_foldEm(const float* __restrict__ H, const float* __restrict__ em,
                         const float* __restrict__ p1w, int midx, int L,
                         float* __restrict__ WtotT) {
  int s = blockIdx.x;        // 0..L-1
  int p = threadIdx.x;       // 0..95
  const float* h  = H + s * NHIP;
  const float* er = em + (size_t)(L - PRED + p) * NHIP;
  float acc = 0.0f;
#pragma unroll 8
  for (int o = 0; o < NHIP; ++o) acc = fmaf(h[o], er[o], acc);
  WtotT[p * SEQ + (SEQ - L + s)] += p1w[midx] * acc;
}

// --------- final channel projection: out[b,p,o] = sum_e pre[be,p]*W2[o,e]+b2

__global__ void k_proj2(const float* __restrict__ pre, const float* __restrict__ w2,
                        const float* __restrict__ b2, float* __restrict__ out) {
  int bp = blockIdx.x;           // b*96+p
  int b = bp / PRED, p = bp % PRED;
  int o = threadIdx.x;           // 0..63
  float acc = b2[o];
#pragma unroll 8
  for (int e = 0; e < 64; ++e)
    acc = fmaf(pre[(size_t)(b * 64 + e) * PRED + p], w2[o * 64 + e], acc);
  out[(size_t)bp * 64 + o] = acc;
}

// ------------------------------- launcher ----------------------------------

extern "C" void kernel_launch(void* const* d_in, const int* in_sizes, int n_in,
                              void* d_out, int out_size, void* d_ws, size_t ws_size,
                              hipStream_t stream) {
  const float* x_enc = (const float*)d_in[0];
  const float* aw    = (const float*)d_in[1];
  const float* ab    = (const float*)d_in[2];
  const float* Am[3] = { (const float*)d_in[3],  (const float*)d_in[8],  (const float*)d_in[13] };
  const float* Bm[3] = { (const float*)d_in[4],  (const float*)d_in[9],  (const float*)d_in[14] };
  const float* em[3] = { (const float*)d_in[5],  (const float*)d_in[10], (const float*)d_in[15] };
  const float* wr[3] = { (const float*)d_in[6],  (const float*)d_in[11], (const float*)d_in[16] };
  const float* wi[3] = { (const float*)d_in[7],  (const float*)d_in[12], (const float*)d_in[17] };
  const float* p1w = (const float*)d_in[18];
  const float* p1b = (const float*)d_in[19];
  const float* p2w = (const float*)d_in[20];
  const float* p2b = (const float*)d_in[21];
  float* out = (float*)d_out;

  char* ws = (char*)d_ws;
  size_t off = 0;
  auto carve = [&](size_t bytes) -> void* {
    void* p = ws + off;
    off = (off + bytes + 255) & ~(size_t)255;
    return p;
  };
  bf16*  XThi   = (bf16*)carve((size_t)BE * SEQ * 2);
  bf16*  XTlo   = (bf16*)carve((size_t)BE * SEQ * 2);
  float* Kbuf   = (float*)carve((size_t)SEQ * NHIP * 4);
  bf16*  ARShi  = (bf16*)carve((size_t)SEQ * KDIM_H * 2);
  bf16*  ARSlo  = (bf16*)carve((size_t)SEQ * KDIM_H * 2);
  bf16*  BpThi  = (bf16*)carve((size_t)NHIP * KDIM_H * 2);
  bf16*  BpTlo  = (bf16*)carve((size_t)NHIP * KDIM_H * 2);
  float* Hbuf   = (float*)carve((size_t)SEQ * NHIP * 4);
  float* WtotT  = (float*)carve((size_t)PRED * SEQ * 4);
  bf16*  Whi    = (bf16*)carve((size_t)PRED * SEQ * 2);
  bf16*  Wlo    = (bf16*)carve((size_t)PRED * SEQ * 2);
  float* pre    = (float*)carve((size_t)BE * PRED * 4);
  (void)ws_size; (void)in_sizes; (void)n_in; (void)out_size;

  // 0) zero the accumulated decode weight matrix
  k_zero_f32<<<(PRED * SEQ + 255) / 256, 256, 0, stream>>>(WtotT, PRED * SEQ);

  // 1) affine + transpose input once (shared by all scales via Wtot)
  k_affine<<<dim3(8, SEQ), 64, 0, stream>>>(x_enc, aw, ab, XThi, XTlo);

  // 2) per-scale: K chain -> R/S correlations -> H GEMM (TDM-staged) -> fold
  const int Ls[3] = { 96, 192, 384 };
  for (int m = 0; m < 3; ++m) {
    int L = Ls[m];
    k_hippoK<<<1, 1024, 0, stream>>>(Am[m], Bm[m], Kbuf, L);
    k_packW <<<256, 256, 0, stream>>>(wr[m], wi[m], BpThi, BpTlo);
    k_RS    <<<L, 256, 0, stream>>>(Kbuf, ARShi, ARSlo, L);
    k_gemmH_lds<<<dim3(NHIP / 32, L / 32), 128, 0, stream>>>(
        ARShi, ARSlo, BpThi, BpTlo, Hbuf, KDIM_H);
    k_foldEm<<<L, PRED, 0, stream>>>(Hbuf, em[m], p1w, m, L, WtotT);
  }

  // 3) apply collapsed weights: pre[be,p] = sum_t XT[be,t]*Wtot[t,p] + p1b
  k_split_bf16<<<(PRED * SEQ + 255) / 256, 256, 0, stream>>>(WtotT, Whi, Wlo, PRED * SEQ);
  k_gemm_bf16x3<<<dim3(PRED / 16, BE / 16), 32, 0, stream>>>(
      XThi, XTlo, Whi, Wlo, pre, SEQ, PRED, p1b, 1);

  // 4) final channel projection
  k_proj2<<<8 * PRED, 64, 0, stream>>>(pre, p2w, p2b, out);
}
